// CausalSelfAttention_34626026341071
// MI455X (gfx1250) — matrix operand order repaired and checked by reading, not compiled, not run on hardware
//
#include <hip/hip_runtime.h>

typedef unsigned short u16;
typedef __attribute__((ext_vector_type(16))) __bf16 v16bf;
typedef __attribute__((ext_vector_type(8)))  float  v8f;

union FragAB { v16bf v; uint4 q[2]; };

__device__ __forceinline__ u16 f2bf(float f) {
  union { __bf16 b; u16 s; } c;
  c.b = (__bf16)f;                 // hardware f32->bf16 convert
  return c.s;
}

#define WMMA_BF16(a, b, c) \
  __builtin_amdgcn_wmma_f32_16x16x32_bf16(false, (a), false, (b), (short)0, (c), false, false)

// ---- CDNA5 async memory->LDS copy (ASYNCcnt path), guarded fallback ----
#if __has_builtin(__builtin_amdgcn_global_load_async_to_lds_b128)
#define HAVE_ASYNC_LDS 1
#else
#define HAVE_ASYNC_LDS 0
#endif

typedef int v4i_vs __attribute__((vector_size(16)));
typedef __attribute__((address_space(1))) v4i_vs* gptr_b128;
typedef __attribute__((address_space(3))) v4i_vs* lptr_b128;

__device__ __forceinline__ void async_cp16(void* lds, const void* glb) {
#if HAVE_ASYNC_LDS
  __builtin_amdgcn_global_load_async_to_lds_b128(
      (gptr_b128)glb, (lptr_b128)lds, 0, 0);
#else
  *(uint4*)lds = *(const uint4*)glb;
#endif
}

__device__ __forceinline__ void async_wait_all() {
#if HAVE_ASYNC_LDS
#if __has_builtin(__builtin_amdgcn_s_wait_asynccnt)
  __builtin_amdgcn_s_wait_asynccnt(0);
#else
  asm volatile("s_wait_asynccnt 0" ::: "memory");
#endif
#endif
}

__device__ __forceinline__ void sched_fence() {
#if __has_builtin(__builtin_amdgcn_sched_barrier)
  __builtin_amdgcn_sched_barrier(0);
#endif
}

constexpr int Bn = 2, Tn = 2048, Cn = 2048, Hn = 16, Dn = 128;

// ---------------------------------------------------------------------------
// f32 -> bf16 pre-conversion (vectorized 4-wide)
// ---------------------------------------------------------------------------
__global__ __launch_bounds__(256) void cvt_f32_bf16(
    const float* __restrict__ in, u16* __restrict__ out, int n)
{
  int i = (blockIdx.x * 256 + threadIdx.x) * 4;
  if (i + 3 < n) {
    float4 f = *(const float4*)(in + i);
    union { u16 s[4]; uint2 u; } o;
    o.s[0] = f2bf(f.x); o.s[1] = f2bf(f.y);
    o.s[2] = f2bf(f.z); o.s[3] = f2bf(f.w);
    *(uint2*)(out + i) = o.u;
  }
}

// ---------------------------------------------------------------------------
// C[M,N] = (A[M,K] @ W[N,K]^T) * outScale, bf16 in, f32 accumulate.
// 256 threads (8 waves), 128x128 tile, K-step 32.
// Wave = 64x32 sub-tile -> 8 WMMAs per K-step.
// Double-buffered async b128 staging: prefetch t+1, compute t, wait, barrier.
// ---------------------------------------------------------------------------
template <bool OUT_BF16>
__global__ __launch_bounds__(256) void gemm_bf16(
    const u16* __restrict__ A, const u16* __restrict__ W,
    void* __restrict__ Cout, int M, int N, int K, float outScale)
{
  __shared__ __align__(16) u16 As[2][128][40];   // 2 x 10 KB
  __shared__ __align__(16) u16 Bs[2][128][40];

  const int tid  = threadIdx.x;
  const int wave = tid >> 5, lane = tid & 31;
  const int wr = wave >> 2;          // 0..1 : M half (64 rows)
  const int wc = wave & 3;           // 0..3 : N quarter (32 cols)
  const int lr = lane & 15, lh = lane >> 4;
  const int m0 = blockIdx.y * 128, n0 = blockIdx.x * 128;

  v8f acc[4][2];
  #pragma unroll
  for (int mi = 0; mi < 4; ++mi)
    #pragma unroll
    for (int ni = 0; ni < 2; ++ni)
      acc[mi][ni] = v8f{0.f, 0.f, 0.f, 0.f, 0.f, 0.f, 0.f, 0.f};

  const int srow = tid >> 1;         // 0..127
  const int sc   = (tid & 1) * 16;   // 0 or 16

  auto stage = [&](int buf, int k0) {
    async_cp16(&As[buf][srow][sc],     A + (size_t)(m0 + srow) * K + k0 + sc);
    async_cp16(&As[buf][srow][sc + 8], A + (size_t)(m0 + srow) * K + k0 + sc + 8);
    async_cp16(&Bs[buf][srow][sc],     W + (size_t)(n0 + srow) * K + k0 + sc);
    async_cp16(&Bs[buf][srow][sc + 8], W + (size_t)(n0 + srow) * K + k0 + sc + 8);
  };

  stage(0, 0);
  async_wait_all();
  __syncthreads();

  const int nk = K / 32;
  for (int t = 0; t < nk; ++t) {
    const int buf = t & 1;
    if (t + 1 < nk) stage(buf ^ 1, (t + 1) * 32);   // prefetch next tile

    FragAB a[4], b[2];
    #pragma unroll
    for (int mi = 0; mi < 4; ++mi) {
      int row = wr * 64 + mi * 16 + lr;
      int kh  = lh * 8;
      a[mi].q[0] = *(const uint4*)&As[buf][row][kh];
      a[mi].q[1] = *(const uint4*)&As[buf][row][kh + 16];
    }
    #pragma unroll
    for (int ni = 0; ni < 2; ++ni) {
      int nr = wc * 32 + ni * 16 + lr;
      int kb = lh * 8;
      b[ni].q[0] = *(const uint4*)&Bs[buf][nr][kb];
      b[ni].q[1] = *(const uint4*)&Bs[buf][nr][kb + 16];
    }
    sched_fence();   // issue all ds_loads before the WMMA chain
    #pragma unroll
    for (int mi = 0; mi < 4; ++mi)
      #pragma unroll
      for (int ni = 0; ni < 2; ++ni)
        acc[mi][ni] = WMMA_BF16(a[mi].v, b[ni].v, acc[mi][ni]);

    async_wait_all();   // prefetch latency hidden under the WMMAs above
    __syncthreads();
  }

  #pragma unroll
  for (int mi = 0; mi < 4; ++mi)
    #pragma unroll
    for (int ni = 0; ni < 2; ++ni)
      #pragma unroll
      for (int r = 0; r < 8; ++r) {
        int row = m0 + wr * 64 + mi * 16 + r + 8 * lh;
        int col = n0 + wc * 32 + ni * 16 + lr;
        float v = acc[mi][ni][r] * outScale;
        if (OUT_BF16) ((u16*)Cout)[(size_t)row * N + col] = f2bf(v);
        else          ((float*)Cout)[(size_t)row * N + col] = v;
      }
}

// ---------------------------------------------------------------------------
// V [B,T,H*D] -> VT [B,H,D,T]  (LDS-tiled 64x64 transpose, one pass)
// ---------------------------------------------------------------------------
__global__ __launch_bounds__(256) void transpose_v(
    const u16* __restrict__ V, u16* __restrict__ VT)
{
  __shared__ __align__(16) u16 Ld[64][72];
  const int tid = threadIdx.x;
  const int t0 = blockIdx.x * 64;
  const int d0 = (blockIdx.y & 1) * 64;
  const int bh = blockIdx.y >> 1;
  const int b = bh / Hn, h = bh % Hn;

  for (int i = tid; i < 512; i += 256) {
    int r = i >> 3, c = (i & 7) * 8;
    *(uint4*)&Ld[r][c] =
        *(const uint4*)(V + ((size_t)(b * Tn + t0 + r) * Cn + h * Dn + d0 + c));
  }
  __syncthreads();
  for (int i = tid; i < 512; i += 256) {
    int d = i >> 3, c = (i & 7) * 8;
    union { uint4 u; u16 s[8]; } o;
    #pragma unroll
    for (int j = 0; j < 8; ++j) o.s[j] = Ld[c + j][d];
    *(uint4*)(VT + ((size_t)((b * Hn + h) * Dn + d0 + d) * Tn + t0 + c)) = o.u;
  }
}

// ---------------------------------------------------------------------------
// Flash attention: block = 64 query rows of one (b,h); 4 waves x 16 rows.
// Q is pre-scaled by 1/sqrt(D). Double-buffered async staging of K and VT.
// ---------------------------------------------------------------------------
__global__ __launch_bounds__(128) void flash_attn(
    const u16* __restrict__ Q, const u16* __restrict__ K,
    const u16* __restrict__ VT, u16* __restrict__ O)
{
  __shared__ __align__(16) u16 Ks[2][32][136];   // [key][d]
  __shared__ __align__(16) u16 Vt[2][128][40];   // [d][key]
  __shared__ __align__(16) u16 Ps[4][16][40];    // per-wave P round-trip

  const int tid  = threadIdx.x;
  const int wave = tid >> 5, lane = tid & 31;
  const int lr = lane & 15, lh = lane >> 4;
  const int qt = blockIdx.x;
  const int bh = blockIdx.y;
  const int b = bh / Hn, h = bh % Hn;

  FragAB qf[4];
  {
    int t = qt * 64 + wave * 16 + lr;
    const u16* qrow = Q + ((size_t)(b * Tn + t) * Cn + h * Dn);
    #pragma unroll
    for (int kk = 0; kk < 4; ++kk) {
      int kh = kk * 32 + lh * 8;
      qf[kk].q[0] = *(const uint4*)(qrow + kh);
      qf[kk].q[1] = *(const uint4*)(qrow + kh + 16);
    }
  }

  v8f o_acc[8];
  float m_run[8], l_run[8];
  #pragma unroll
  for (int dt = 0; dt < 8; ++dt)
    o_acc[dt] = v8f{0.f, 0.f, 0.f, 0.f, 0.f, 0.f, 0.f, 0.f};
  #pragma unroll
  for (int r = 0; r < 8; ++r) { m_run[r] = -1e30f; l_run[r] = 0.f; }

  auto stage_kv = [&](int buf, int jt) {
    for (int i = tid; i < 512; i += 128) {
      int kr = i >> 4, c8 = (i & 15) * 8;
      async_cp16(&Ks[buf][kr][c8],
                 K + ((size_t)(b * Tn + jt * 32 + kr) * Cn + h * Dn + c8));
    }
    for (int i = tid; i < 512; i += 128) {
      int dr = i >> 2, c8 = (i & 3) * 8;
      async_cp16(&Vt[buf][dr][c8],
                 VT + ((size_t)((b * Hn + h) * Dn + dr) * Tn + jt * 32 + c8));
    }
  };

  const int njt = 2 * (qt + 1);
  stage_kv(0, 0);
  async_wait_all();
  __syncthreads();

  for (int jt = 0; jt < njt; ++jt) {
    const int buf = jt & 1;
    if (jt + 1 < njt) stage_kv(buf ^ 1, jt + 1);   // prefetch next key tile

    // ---- S = Q K^T: preload all 8 K fragments, then 8 WMMAs ----
    FragAB kf[8];
    #pragma unroll
    for (int kk = 0; kk < 4; ++kk) {
      int db = kk * 32 + lh * 8;
      kf[2 * kk + 0].q[0] = *(const uint4*)&Ks[buf][lr][db];
      kf[2 * kk + 0].q[1] = *(const uint4*)&Ks[buf][lr][db + 16];
      kf[2 * kk + 1].q[0] = *(const uint4*)&Ks[buf][16 + lr][db];
      kf[2 * kk + 1].q[1] = *(const uint4*)&Ks[buf][16 + lr][db + 16];
    }
    sched_fence();
    v8f s0 = v8f{0.f,0.f,0.f,0.f,0.f,0.f,0.f,0.f};
    v8f s1 = v8f{0.f,0.f,0.f,0.f,0.f,0.f,0.f,0.f};
    #pragma unroll
    for (int kk = 0; kk < 4; ++kk) {
      s0 = WMMA_BF16(qf[kk].v, kf[2 * kk + 0].v, s0);
      s1 = WMMA_BF16(qf[kk].v, kf[2 * kk + 1].v, s1);
    }

    // ---- online softmax (Q pre-scaled; row = r + 8*lh) ----
    #pragma unroll
    for (int r = 0; r < 8; ++r) {
      int qrow = qt * 64 + wave * 16 + r + 8 * lh;
      int key0 = jt * 32 + lr;
      int key1 = key0 + 16;
      float e0 = s0[r]; if (key0 > qrow) e0 = -1e30f;
      float e1 = s1[r]; if (key1 > qrow) e1 = -1e30f;
      float mx = fmaxf(e0, e1);
      mx = fmaxf(mx, __shfl_xor(mx, 1, 32));
      mx = fmaxf(mx, __shfl_xor(mx, 2, 32));
      mx = fmaxf(mx, __shfl_xor(mx, 4, 32));
      mx = fmaxf(mx, __shfl_xor(mx, 8, 32));
      float mnew  = fmaxf(m_run[r], mx);
      float alpha = __expf(m_run[r] - mnew);
      float p0 = __expf(e0 - mnew);
      float p1 = __expf(e1 - mnew);
      float rs = p0 + p1;
      rs += __shfl_xor(rs, 1, 32);
      rs += __shfl_xor(rs, 2, 32);
      rs += __shfl_xor(rs, 4, 32);
      rs += __shfl_xor(rs, 8, 32);
      l_run[r] = l_run[r] * alpha + rs;
      m_run[r] = mnew;
      #pragma unroll
      for (int dt = 0; dt < 8; ++dt) o_acc[dt][r] *= alpha;
      Ps[wave][r + 8 * lh][lr]      = f2bf(p0);
      Ps[wave][r + 8 * lh][16 + lr] = f2bf(p1);
    }

    // ---- P: C-layout -> A-fragment via wave-private LDS (DS in-order) ----
    FragAB pa;
    {
      int kh = lh * 8;
      pa.q[0] = *(const uint4*)&Ps[wave][lr][kh];
      pa.q[1] = *(const uint4*)&Ps[wave][lr][kh + 16];
    }

    // ---- O += P @ V: preload all 8 V fragments, then 8 WMMAs ----
    FragAB vf[8];
    #pragma unroll
    for (int dt = 0; dt < 8; ++dt) {
      int d = dt * 16 + lr, kb = lh * 8;
      vf[dt].q[0] = *(const uint4*)&Vt[buf][d][kb];
      vf[dt].q[1] = *(const uint4*)&Vt[buf][d][kb + 16];
    }
    sched_fence();
    #pragma unroll
    for (int dt = 0; dt < 8; ++dt)
      o_acc[dt] = WMMA_BF16(pa.v, vf[dt].v, o_acc[dt]);

    async_wait_all();   // prefetch latency hidden under compute above
    __syncthreads();
  }

  #pragma unroll
  for (int r = 0; r < 8; ++r) {
    float inv = 1.0f / l_run[r];
    int t = qt * 64 + wave * 16 + r + 8 * lh;
    u16* orow = O + ((size_t)(b * Tn + t) * Cn + h * Dn);
    #pragma unroll
    for (int dt = 0; dt < 8; ++dt)
      orow[dt * 16 + lr] = f2bf(o_acc[dt][r] * inv);
  }
}

// ---------------------------------------------------------------------------
extern "C" void kernel_launch(void* const* d_in, const int* in_sizes, int n_in,
                              void* d_out, int out_size, void* d_ws, size_t ws_size,
                              hipStream_t stream) {
  const float* x  = (const float*)d_in[0];
  const float* Wq = (const float*)d_in[1];
  const float* Wk = (const float*)d_in[2];
  const float* Wv = (const float*)d_in[3];
  const float* Wo = (const float*)d_in[4];

  const size_t NX = (size_t)Bn * Tn * Cn;   // 8,388,608
  const size_t NW = (size_t)Cn * Cn;        // 4,194,304

  u16* xb  = (u16*)d_ws;
  u16* wqb = xb  + NX;
  u16* wkb = wqb + NW;
  u16* wvb = wkb + NW;
  u16* wob = wvb + NW;
  u16* Qb  = wob + NW;
  u16* Kb  = Qb  + NX;
  u16* Vb  = Kb  + NX;
  u16* VTb = Vb  + NX;
  u16* Ab  = VTb + NX;

  cvt_f32_bf16<<<(NX / 4 + 255) / 256, 256, 0, stream>>>(x,  xb,  (int)NX);
  cvt_f32_bf16<<<(NW / 4 + 255) / 256, 256, 0, stream>>>(Wq, wqb, (int)NW);
  cvt_f32_bf16<<<(NW / 4 + 255) / 256, 256, 0, stream>>>(Wk, wkb, (int)NW);
  cvt_f32_bf16<<<(NW / 4 + 255) / 256, 256, 0, stream>>>(Wv, wvb, (int)NW);
  cvt_f32_bf16<<<(NW / 4 + 255) / 256, 256, 0, stream>>>(Wo, wob, (int)NW);

  const int M = Bn * Tn;                    // 4096
  const float qscale = 0.08838834764831845f;  // 1/sqrt(128)
  dim3 gg(Cn / 128, M / 128);               // (16, 32)
  gemm_bf16<true><<<gg, 256, 0, stream>>>(xb, wqb, Qb, M, Cn, Cn, qscale);
  gemm_bf16<true><<<gg, 256, 0, stream>>>(xb, wkb, Kb, M, Cn, Cn, 1.0f);
  gemm_bf16<true><<<gg, 256, 0, stream>>>(xb, wvb, Vb, M, Cn, Cn, 1.0f);

  dim3 gt(Tn / 64, 2 * Bn * Hn);            // (32, 64)
  transpose_v<<<gt, 256, 0, stream>>>(Vb, VTb);

  dim3 ga(Tn / 64, Bn * Hn);                // (32, 32)
  flash_attn<<<ga, 128, 0, stream>>>(Qb, Kb, VTb, Ab);

  gemm_bf16<false><<<gg, 256, 0, stream>>>(Ab, wob, d_out, M, Cn, Cn, 1.0f);
}
